// IRMC_GC_Model_1906965479724
// MI455X (gfx1250) — matrix operand order repaired and checked by reading, not compiled, not run on hardware
//
#include <hip/hip_runtime.h>
#include <hip/hip_bf16.h>

typedef __attribute__((ext_vector_type(16))) __bf16 v16bf;
typedef __attribute__((ext_vector_type(8)))  __bf16 v8bf;
typedef __attribute__((ext_vector_type(8)))  float  v8f;

#define KB 1024   // batch
#define KL 50     // history length
#define KS 256    // samples
#define KH 4      // heads
#define KE 64     // embedding

// ---- fast tanh: CDNA5 hardware transcendental ------------------------------
__device__ __forceinline__ float fast_tanh(float x) {
    float r;
    asm volatile("v_tanh_f32 %0, %1" : "=v"(r) : "v"(x));
    return r;
}

// ---- WMMA fragment helpers (wave32, 16x16x32 bf16) -------------------------
// A: 16x32 (MxK) row-major in LDS. Lane layout: m = lane&15, half = lane>>4.
// Lane's 16 elems = k in [8h, 8h+8) then [16+8h, 16+8h+8): two 16-byte loads.
template<int STRIDE>
__device__ __forceinline__ v16bf load_frag_A(const __bf16* A, int m0, int k0, int lane) {
    int m = m0 + (lane & 15);
    int half = lane >> 4;
    const __bf16* base = A + m * STRIDE + k0 + 8 * half;
    v8bf lo = *(const v8bf*)(base);
    v8bf hi = *(const v8bf*)(base + 16);
    return __builtin_shufflevector(lo, hi, 0, 1, 2, 3, 4, 5, 6, 7,
                                           8, 9, 10, 11, 12, 13, 14, 15);
}

// B: 32x16 (KxN) tile read from LDS staged TRANSPOSED (Bt[n][k], stride=K).
// Lane holds column n = lane&15; k in [k0+8*half, +8) then [k0+16+8*half, +8).
template<int STRIDE>
__device__ __forceinline__ v16bf load_frag_BT(const __bf16* Bt, int k0, int n0, int lane) {
    int n = n0 + (lane & 15);
    int half = lane >> 4;
    const __bf16* base = Bt + n * STRIDE + k0 + 8 * half;
    v8bf lo = *(const v8bf*)(base);
    v8bf hi = *(const v8bf*)(base + 16);
    return __builtin_shufflevector(lo, hi, 0, 1, 2, 3, 4, 5, 6, 7,
                                           8, 9, 10, 11, 12, 13, 14, 15);
}

__device__ __forceinline__ v8f wmma_bf16(v16bf a, v16bf b, v8f c) {
    return __builtin_amdgcn_wmma_f32_16x16x32_bf16(
        /*neg_a=*/false, a, /*neg_b=*/false, b,
        /*c_mod=*/(short)0, c, /*reuse_a=*/false, /*reuse_b=*/false);
}

// C/D: lane holds col n=lane&15; VGPR i -> row m = i + 8*(lane>>4).
template<int STRIDE>
__device__ __forceinline__ void store_frag_C_bf16(__bf16* C, int m0, int n0, int lane, v8f c) {
    int n = n0 + (lane & 15);
    int half = lane >> 4;
#pragma unroll
    for (int i = 0; i < 8; ++i)
        C[(m0 + i + 8 * half) * STRIDE + n] = (__bf16)c[i];
}

// ---- Kernel 1: user_fea_encode --------------------------------------------
__global__ __launch_bounds__(128)
void fea_encode_kernel(const int* __restrict__ x, const int* __restrict__ his,
                       const int* __restrict__ hl,
                       const float* __restrict__ user_table,
                       const float* __restrict__ item_emb,
                       const float* __restrict__ W_att_w,
                       const float* __restrict__ W_att_b,
                       const float* __restrict__ W_agg_w,
                       float* __restrict__ fea_out) {
    __shared__ alignas(32) __bf16 hist_h[64 * 64];   // 8 KB, row-major (l,e)
    __shared__ alignas(32) __bf16 WbT[64 * 64];      // 8 KB, transposed (n,e)
    __shared__ float  hist_f[64 * 64];               // 16 KB
    __shared__ float  keyt[64 * 64];                 // 16 KB
    __shared__ float  u_s[64];
    __shared__ float  ev[64];
    __shared__ float  inv_den;
    __shared__ float  outv[64];

    const int b = blockIdx.x;
    const int tid = threadIdx.x;
    const int uid = x[2 * b + 0];
    const int len = hl[b];

    if (tid < 64) u_s[tid] = user_table[(long)uid * KE + tid];
    // stage W_att transposed: WbT[n*64 + e] = W_att[e*64 + n]
    for (int i = tid; i < 64 * 64; i += 128) {
        int e = i >> 6, n = i & 63;
        WbT[n * 64 + e] = (__bf16)W_att_w[i];
    }

    for (int i = tid; i < 64 * 64; i += 128) {
        int l = i >> 6, e = i & 63;
        float v = 0.f;
        if (l < KL && l < len) {
            int item = his[b * KL + l];
            v = item_emb[(long)item * KE + e];
        }
        hist_f[i] = v;
        hist_h[i] = (__bf16)v;
    }
    __syncthreads();

    const int wave = tid >> 5, lane = tid & 31;
    // 4 M-tiles x 4 N-tiles, 4 tiles per wave
#pragma unroll
    for (int t = 0; t < 4; ++t) {
        int tile = wave * 4 + t;
        int mt = tile >> 2, nt = tile & 3;
        v8f acc = {};
#pragma unroll
        for (int ks = 0; ks < 2; ++ks) {
            v16bf a  = load_frag_A<64>(hist_h, mt * 16, ks * 32, lane);
            v16bf bb = load_frag_BT<64>(WbT, ks * 32, nt * 16, lane);
            acc = wmma_bf16(a, bb, acc);
        }
        int n = nt * 16 + (lane & 15);
        int half = lane >> 4;
        float bias = W_att_b[n];
#pragma unroll
        for (int i = 0; i < 8; ++i)
            keyt[(mt * 16 + i + 8 * half) * 64 + n] = fast_tanh(acc[i] + bias);
    }
    __syncthreads();

    // att = dot(keyt[l], u); pad rows (hist sum==0) get att=0 -> exp=1 (ref)
    if (tid < KL) {
        float s = 0.f, hs = 0.f;
        for (int e = 0; e < 64; ++e) {
            s += keyt[tid * 64 + e] * u_s[e];
            hs += hist_f[tid * 64 + e];
        }
        float att = (hs == 0.f) ? 0.f : s;
        ev[tid] = __expf(att);
    }
    __syncthreads();
    if (tid == 0) {
        float d = 0.f;
        for (int l = 0; l < KL; ++l) d += ev[l];
        inv_den = 1.f / (d + 1e-12f);
    }
    __syncthreads();
    if (tid < 64) {
        float acc = 0.f;
        for (int l = 0; l < KL; ++l) acc += ev[l] * hist_f[l * 64 + tid];
        outv[tid] = acc * inv_den;
    }
    __syncthreads();
    if (tid < 64) {
        float acc = 0.f;
        for (int e = 0; e < 64; ++e) acc += outv[e] * W_agg_w[e * 64 + tid];
        fea_out[b * KE + tid] = acc;
    }
}

// ---- Kernel 2: get_user_emb attention per (h,b) ----------------------------
__global__ __launch_bounds__(128)
void attn_kernel(const float* __restrict__ fea,
                 const int* __restrict__ sample_idx,   // already offset per call
                 const int* __restrict__ supp_users,
                 const float* __restrict__ tgt_user_emb,
                 const float* __restrict__ Wq,
                 const float* __restrict__ Wk,
                 const float* __restrict__ Wv,
                 float* __restrict__ g_out) {
    __shared__ alignas(32) __bf16 neigh[KS * KE];    // 32 KB, row-major (s,e)
    __shared__ alignas(32) __bf16 WkT[64 * 64];      // 8 KB, transposed (n,e)
    __shared__ alignas(32) __bf16 kmat[KS * KE];     // 32 KB
    __shared__ int    ru[KS];
    __shared__ float  q_s[64];
    __shared__ float  fea_s[64];
    __shared__ float  sc[KS];
    __shared__ float  ctx[64];
    __shared__ float  red[2];

    const int hb = blockIdx.x;
    const int h = hb >> 10;              // KB = 1024
    const int b = hb & (KB - 1);
    const int tid = threadIdx.x;

    if (tid < 64) fea_s[tid] = fea[b * KE + tid];
    for (int i = tid; i < 64 * 64; i += 128) {
        int e = i >> 6, n = i & 63;
        WkT[n * 64 + e] = (__bf16)Wk[h * 4096 + i];
    }

    const int* sidx = sample_idx + ((long)h * KB + b) * KS;
    for (int s = tid; s < KS; s += 128) ru[s] = supp_users[sidx[s]];
    __syncthreads();

    if (tid < 64) {
        float acc = 0.f;
        for (int e = 0; e < 64; ++e) acc += fea_s[e] * Wq[h * 4096 + e * 64 + tid];
        q_s[tid] = acc;
    }
    // gather neighbor rows as float4 and down-convert
    for (int i = tid; i < KS * 16; i += 128) {
        int s = i >> 4, c = (i & 15) * 4;
        const float4 v = *(const float4*)(tgt_user_emb + (long)ru[s] * KE + c);
        neigh[s * KE + c + 0] = (__bf16)v.x;
        neigh[s * KE + c + 1] = (__bf16)v.y;
        neigh[s * KE + c + 2] = (__bf16)v.z;
        neigh[s * KE + c + 3] = (__bf16)v.w;
    }
    __syncthreads();

    const int wave = tid >> 5, lane = tid & 31;
    // k = neigh(256x64) @ Wk(64x64): 16 M-tiles over 4 waves
    for (int mt = wave; mt < 16; mt += 4) {
#pragma unroll
        for (int nt = 0; nt < 4; ++nt) {
            v8f acc = {};
#pragma unroll
            for (int ks = 0; ks < 2; ++ks) {
                v16bf a  = load_frag_A<64>(neigh, mt * 16, ks * 32, lane);
                v16bf bb = load_frag_BT<64>(WkT, ks * 32, nt * 16, lane);
                acc = wmma_bf16(a, bb, acc);
            }
            store_frag_C_bf16<64>(kmat, mt * 16, nt * 16, lane, acc);
        }
    }
    __syncthreads();

    for (int s = tid; s < KS; s += 128) {
        float acc = 0.f;
        for (int o = 0; o < 64; ++o) acc += q_s[o] * (float)kmat[s * KE + o];
        sc[s] = acc;
    }
    __syncthreads();
    if (tid == 0) {
        float m = sc[0];
        for (int s = 1; s < KS; ++s) m = fmaxf(m, sc[s]);
        red[0] = m;
    }
    __syncthreads();
    for (int s = tid; s < KS; s += 128) sc[s] = __expf(sc[s] - red[0]);
    __syncthreads();
    if (tid == 0) {
        float d = 0.f;
        for (int s = 0; s < KS; ++s) d += sc[s];
        red[1] = 1.f / d;
    }
    __syncthreads();
    if (tid < 64) {
        float acc = 0.f;
        for (int s = 0; s < KS; ++s) acc += sc[s] * (float)kmat[s * KE + tid];
        ctx[tid] = acc * red[1];
    }
    __syncthreads();
    if (tid < 64) {
        float acc = 0.f;
        for (int o = 0; o < 64; ++o) acc += ctx[o] * Wv[h * 4096 + o * 64 + tid];
        g_out[(long)b * (KH * KE) + h * KE + tid] = acc;
    }
}

// ---- Kernel 3: (B,256) @ W_out(256,64) via WMMA ----------------------------
__global__ __launch_bounds__(128)
void out_proj_kernel(const float* __restrict__ A, const float* __restrict__ W,
                     float* __restrict__ out) {
    __shared__ alignas(32) __bf16 As[64 * 256];   // 32 KB, row-major (m,k)
    __shared__ alignas(32) __bf16 WsT[64 * 256];  // 32 KB, transposed (n,k)
    const int row0 = blockIdx.x * 64;
    const int tid = threadIdx.x;
    for (int i = tid; i < 64 * 256; i += 128) As[i] = (__bf16)A[(long)row0 * 256 + i];
    for (int i = tid; i < 256 * 64; i += 128) {
        int k = i >> 6, n = i & 63;
        WsT[n * 256 + k] = (__bf16)W[i];
    }
    __syncthreads();

    const int wave = tid >> 5, lane = tid & 31;
    const int mt = wave;   // one 16-row M-tile per wave
#pragma unroll
    for (int nt = 0; nt < 4; ++nt) {
        v8f acc = {};
#pragma unroll
        for (int ks = 0; ks < 8; ++ks) {
            v16bf a  = load_frag_A<256>(As, mt * 16, ks * 32, lane);
            v16bf bb = load_frag_BT<256>(WsT, ks * 32, nt * 16, lane);
            acc = wmma_bf16(a, bb, acc);
        }
        int n = nt * 16 + (lane & 15);
        int half = lane >> 4;
#pragma unroll
        for (int i = 0; i < 8; ++i)
            out[(long)(row0 + mt * 16 + i + 8 * half) * 64 + n] = acc[i];
    }
}

// ---- Kernel 4: head (elementwise + tiny MLP) -------------------------------
__global__ __launch_bounds__(128)
void head_kernel(const int* __restrict__ x,
                 const float* __restrict__ user_emb,
                 const float* __restrict__ hyb_emb,
                 const float* __restrict__ tgt_item_emb,
                 const float* __restrict__ l1_w, const float* __restrict__ l1_b,
                 const float* __restrict__ l2_w, const float* __restrict__ l2_b,
                 const float* __restrict__ l3_w, const float* __restrict__ l3_b,
                 float* __restrict__ out) {
    __shared__ float xs[2][128];
    __shared__ float x1[2][64];
    __shared__ float x2s[2][64];
    __shared__ float rsum[2][64];

    const int tid = threadIdx.x;
    const int r = tid >> 6, c = tid & 63;
    const int b = blockIdx.x * 2 + r;
    const int item = x[2 * b + 1];

    const float ie = tgt_item_emb[(long)item * KE + c];
    const float ue = user_emb[b * KE + c];
    const float oe = ue * ie;
    // d_out layout: [0,B) output | [B,2B) x3 | [2B,2B+B*64) out_emb_s | then x2
    out[2 * KB + (long)b * KE + c] = oe;
    rsum[r][c] = oe;
    xs[r][c] = hyb_emb[b * KE + c];
    xs[r][64 + c] = ie;
    __syncthreads();

    if (c == 0) {
        float s = 0.f;
        for (int i = 0; i < 64; ++i) s += rsum[r][i];
        out[b] = s;
    }
    float acc = l1_b[c];
    for (int i = 0; i < 128; ++i) acc += xs[r][i] * l1_w[i * 64 + c];
    x1[r][c] = fast_tanh(acc);
    __syncthreads();

    float acc2 = l2_b[c];
    for (int i = 0; i < 64; ++i) acc2 += x1[r][i] * l2_w[i * 64 + c];
    float v2 = fast_tanh(acc2);
    x2s[r][c] = v2;
    out[2 * KB + KB * KE + (long)b * KE + c] = v2;
    __syncthreads();

    if (c == 0) {
        float s = l3_b[0];
        for (int i = 0; i < 64; ++i) s += x2s[r][i] * l3_w[i];
        out[KB + b] = s;
    }
}

extern "C" void kernel_launch(void* const* d_in, const int* in_sizes, int n_in,
                              void* d_out, int out_size, void* d_ws, size_t ws_size,
                              hipStream_t stream) {
    (void)in_sizes; (void)n_in; (void)out_size; (void)ws_size;
    const int*   x            = (const int*)d_in[0];
    const int*   src_his      = (const int*)d_in[1];
    const int*   src_hl       = (const int*)d_in[2];
    const int*   tgt_his      = (const int*)d_in[3];
    const int*   tgt_hl       = (const int*)d_in[4];
    const int*   sample_idx   = (const int*)d_in[5];
    const int*   supp_users   = (const int*)d_in[6];
    const float* src_user_emb = (const float*)d_in[7];
    const float* src_item_emb = (const float*)d_in[8];
    const float* tgt_user_emb = (const float*)d_in[9];
    const float* tgt_item_emb = (const float*)d_in[10];
    const float* W_att_w      = (const float*)d_in[11];
    const float* W_att_b      = (const float*)d_in[12];
    const float* W_agg_w      = (const float*)d_in[13];
    const float* Wq           = (const float*)d_in[14];
    const float* Wk           = (const float*)d_in[15];
    const float* Wv           = (const float*)d_in[16];
    const float* W_out        = (const float*)d_in[17];
    const float* l1_w         = (const float*)d_in[18];
    const float* l1_b         = (const float*)d_in[19];
    const float* l2_w         = (const float*)d_in[20];
    const float* l2_b         = (const float*)d_in[21];
    const float* l3_w         = (const float*)d_in[22];
    const float* l3_b         = (const float*)d_in[23];
    float* out = (float*)d_out;

    float* fea_src = (float*)d_ws;                 // B*64
    float* fea_tgt = fea_src + KB * KE;            // B*64
    float* g_src   = fea_tgt + KB * KE;            // B*256
    float* g_tgt   = g_src + KB * KH * KE;         // B*256
    float* uemb    = g_tgt + KB * KH * KE;         // B*64
    float* hemb    = uemb + KB * KE;               // B*64

    // Note: reference's user_fea_encode gathers from src_item_emb in BOTH calls.
    fea_encode_kernel<<<KB, 128, 0, stream>>>(x, src_his, src_hl, src_user_emb,
                                              src_item_emb, W_att_w, W_att_b,
                                              W_agg_w, fea_src);
    fea_encode_kernel<<<KB, 128, 0, stream>>>(x, tgt_his, tgt_hl, tgt_user_emb,
                                              src_item_emb, W_att_w, W_att_b,
                                              W_agg_w, fea_tgt);

    attn_kernel<<<KH * KB, 128, 0, stream>>>(fea_src, sample_idx,
                                             supp_users, tgt_user_emb,
                                             Wq, Wk, Wv, g_src);
    attn_kernel<<<KH * KB, 128, 0, stream>>>(fea_tgt, sample_idx + KH * KB * KS,
                                             supp_users, tgt_user_emb,
                                             Wq, Wk, Wv, g_tgt);

    out_proj_kernel<<<KB / 64, 128, 0, stream>>>(g_src, W_out, uemb);
    out_proj_kernel<<<KB / 64, 128, 0, stream>>>(g_tgt, W_out, hemb);

    head_kernel<<<KB / 2, 128, 0, stream>>>(x, uemb, hemb, tgt_item_emb,
                                            l1_w, l1_b, l2_w, l2_b, l3_w, l3_b,
                                            out);
}
